// Network_60146722013662
// MI455X (gfx1250) — matrix operand order repaired and checked by reading, not compiled, not run on hardware
//
#include <hip/hip_runtime.h>

typedef __attribute__((ext_vector_type(16))) _Float16 v16h;
typedef __attribute__((ext_vector_type(8)))  _Float16 v8h;
typedef __attribute__((ext_vector_type(8)))  float    v8f;

#define B_    4096
#define N0_   10
#define N1_   4096
#define N2_   784
#define KP_   800            // N2_ padded up to a multiple of 32 (zero fill)
#define T_    20
#define NTH2  512
#define NW2   (NTH2 / 32)

// ---------------------------------------------------------------------------
// Kernel 0: convert x [B_,784] and W1 [N1_,784] to f16 rows padded to KP_=800.
// B_ == N1_ so one block handles row r of both.
// ---------------------------------------------------------------------------
__global__ __launch_bounds__(256) void convert_f16(
    const float* __restrict__ x, const float* __restrict__ w,
    _Float16* __restrict__ xh, _Float16* __restrict__ wh)
{
  const int r = blockIdx.x;
  const float* __restrict__ xr = x + (size_t)r * N2_;
  const float* __restrict__ wr = w + (size_t)r * N2_;
  _Float16* __restrict__ xo = xh + (size_t)r * KP_;
  _Float16* __restrict__ wo = wh + (size_t)r * KP_;
  for (int k = threadIdx.x; k < KP_; k += 256) {
    float vx = (k < N2_) ? xr[k] : 0.f;
    float vw = (k < N2_) ? wr[k] : 0.f;
    xo[k] = (_Float16)vx;
    wo[k] = (_Float16)vw;
  }
}

// ---------------------------------------------------------------------------
// Kernel 1: xW1[b][j] = sum_k x[b][k]*W1[j][k] + b1[j], f16 WMMA, f32 accum.
// One 32x32 output block per wave (2x2 tiles): 8 vector loads feed 4 WMMAs,
// no converts and no predication in the K loop (K pre-padded to 800).
// ---------------------------------------------------------------------------
__global__ __launch_bounds__(256) void xw1_gemm_f16(
    const _Float16* __restrict__ xh,  // [B_,  KP_] zero-padded
    const _Float16* __restrict__ wh,  // [N1_, KP_] zero-padded
    const float* __restrict__ bias,   // [N1_]
    float* __restrict__ out)          // [B_, N1_]
{
  const int lane  = threadIdx.x & 31;
  const int wave  = threadIdx.x >> 5;
  const int job   = blockIdx.x * 8 + wave;   // 128*128 = 16384 jobs
  const int mbase = (job >> 7) << 5;         // 32-row block base
  const int nbase = (job & 127) << 5;        // 32-col block base
  const int half  = lane >> 4;               // 0 or 1
  const int l15   = lane & 15;
  const int aOfs  = half * 16;               // A 16x32 f16 lane K offset (contiguous k = 16h + s)
  const int bOfs  = half * 16;               // B 32x16 f16 lane K offset (contiguous k = 16h + s)

  const _Float16* pa0 = xh + (size_t)(mbase + l15) * KP_ + aOfs;
  const _Float16* pa1 = pa0 + (size_t)16 * KP_;
  const _Float16* pb0 = wh + (size_t)(nbase + l15) * KP_ + bOfs;
  const _Float16* pb1 = pb0 + (size_t)16 * KP_;

  v8f c00 = {}, c01 = {}, c10 = {}, c11 = {};
  for (int kb = 0; kb < KP_; kb += 32) {
    // A lane data: K = aOfs..aOfs+15 contiguous (matches B)
    v8h a0l = *(const v8h*)(pa0 + kb);
    v8h a0h = *(const v8h*)(pa0 + kb + 8);
    v8h a1l = *(const v8h*)(pa1 + kb);
    v8h a1h = *(const v8h*)(pa1 + kb + 8);
    // B lane data: K = bOfs..bOfs+15 contiguous (32B -> 2x b128)
    v16h b0 = *(const v16h*)(pb0 + kb);
    v16h b1 = *(const v16h*)(pb1 + kb);
    v16h a0 = __builtin_shufflevector(a0l, a0h,
              0,1,2,3,4,5,6,7,8,9,10,11,12,13,14,15);
    v16h a1 = __builtin_shufflevector(a1l, a1h,
              0,1,2,3,4,5,6,7,8,9,10,11,12,13,14,15);
    c00 = __builtin_amdgcn_wmma_f32_16x16x32_f16(false, a0, false, b0,
                                                 (short)0, c00, false, false);
    c01 = __builtin_amdgcn_wmma_f32_16x16x32_f16(false, a0, false, b1,
                                                 (short)0, c01, false, false);
    c10 = __builtin_amdgcn_wmma_f32_16x16x32_f16(false, a1, false, b0,
                                                 (short)0, c10, false, false);
    c11 = __builtin_amdgcn_wmma_f32_16x16x32_f16(false, a1, false, b1,
                                                 (short)0, c11, false, false);
  }

  const int   col0 = nbase + l15;
  const float bi0  = bias[col0];
  const float bi1  = bias[col0 + 16];
  const int   mr   = mbase + half * 8;       // C/D layout: M = vgpr + 8*half
#pragma unroll
  for (int vv = 0; vv < 8; ++vv) {
    out[(size_t)(mr + vv)      * N1_ + col0     ] = c00[vv] + bi0;
    out[(size_t)(mr + vv)      * N1_ + col0 + 16] = c01[vv] + bi1;
    out[(size_t)(mr + 16 + vv) * N1_ + col0     ] = c10[vv] + bi0;
    out[(size_t)(mr + 16 + vv) * N1_ + col0 + 16] = c11[vv] + bi1;
  }
}

// ---------------------------------------------------------------------------
// Kernel 2: fused 20-step EP relaxation. One block per batch row.
// s1/phi1/xW1 live in registers; W0 (160 KB) lives in dynamic LDS; layer-0
// state is replicated per-thread. Only final states ever touch HBM.
// ---------------------------------------------------------------------------
__global__ __launch_bounds__(NTH2) void ep_relax(
    const float* __restrict__ s0_in,  const float* __restrict__ s1_in,
    const float* __restrict__ phi0_in, const float* __restrict__ phi1_in,
    const float* __restrict__ W0,     const float* __restrict__ b0,
    const float* __restrict__ xW1,    // [B_, N1_] = x @ W1.T + b1
    float* __restrict__ o_s0,  float* __restrict__ o_s1,
    float* __restrict__ o_phi0, float* __restrict__ o_phi1)
{
  extern __shared__ float smem[];
  float* W0s = smem;                      // N0_*N1_ = 40960 floats (160 KB)
  float* red = smem + N0_ * N1_;          // NW2*N0_ partials
  float* t0s = red + NW2 * N0_;           // N0_ broadcast slot

  const int row  = blockIdx.x;
  const int tid  = threadIdx.x;
  const int lane = tid & 31;
  const int wv   = tid >> 5;

  for (int i = tid; i < N0_ * N1_; i += NTH2) W0s[i] = W0[i];

  float s1r[8], p1r[8], xwr[8];
#pragma unroll
  for (int i = 0; i < 8; ++i) {
    size_t g = (size_t)row * N1_ + tid + NTH2 * i;
    s1r[i] = s1_in[g]; p1r[i] = phi1_in[g]; xwr[i] = xW1[g];
  }
  float s0r[N0_], p0r[N0_], b0r[N0_];
#pragma unroll
  for (int k = 0; k < N0_; ++k) {
    s0r[k] = s0_in[row * N0_ + k];
    p0r[k] = phi0_in[row * N0_ + k];
    b0r[k] = b0[k];
  }
  __syncthreads();

  for (int step = 0; step < T_; ++step) {
    // ---- t0 = W0 @ bin1 (bin1 from OLD s1,phi1) --------------------------
    float pt[N0_];
#pragma unroll
    for (int k = 0; k < N0_; ++k) pt[k] = 0.f;
#pragma unroll
    for (int i = 0; i < 8; ++i) {
      int j = tid + NTH2 * i;
      float bin1 = (s1r[i] + p1r[i] >= 0.5f) ? 1.f : 0.f;
#pragma unroll
      for (int k = 0; k < N0_; ++k) pt[k] = fmaf(bin1, W0s[k * N1_ + j], pt[k]);
    }
#pragma unroll
    for (int off = 16; off > 0; off >>= 1)
#pragma unroll
      for (int k = 0; k < N0_; ++k) pt[k] += __shfl_down(pt[k], off, 32);
    if (lane == 0)
#pragma unroll
      for (int k = 0; k < N0_; ++k) red[wv * N0_ + k] = pt[k];
    __syncthreads();
    if (tid < N0_) {
      float s = 0.f;
      for (int w = 0; w < NW2; ++w) s += red[w * N0_ + tid];
      t0s[tid] = s;
    }
    __syncthreads();

    // bin0 from OLD s0,phi0 (replicated)
    float bin0[N0_];
#pragma unroll
    for (int k = 0; k < N0_; ++k) bin0[k] = (s0r[k] + p0r[k] >= 0.5f) ? 1.f : 0.f;

    // ---- layer-1 update: pre1 = rhop(s1-.5) * (xW1 + bin0 @ W0) ----------
#pragma unroll
    for (int i = 0; i < 8; ++i) {
      int j = tid + NTH2 * i;
      float a1 = 0.f;
#pragma unroll
      for (int k = 0; k < N0_; ++k) a1 = fmaf(bin0[k], W0s[k * N1_ + j], a1);
      float gate = (s1r[i] >= 0.f && s1r[i] <= 1.f) ? 1.f : 0.f;
      float pre1 = gate * (xwr[i] + a1);
      float s1n  = fminf(fmaxf(0.5f * s1r[i] + 0.5f * pre1, 0.f), 1.f);
      float bn   = (s1n + p1r[i] >= 0.5f) ? 1.f : 0.f;   // uses OLD phi1
      p1r[i] = p1r[i] + s1n - bn;
      s1r[i] = s1n;
    }

    // ---- layer-0 update: pre0 = rhop(s0-.5) * (t0 + b0) ------------------
#pragma unroll
    for (int k = 0; k < N0_; ++k) {
      float gate = (s0r[k] >= 0.f && s0r[k] <= 1.f) ? 1.f : 0.f;
      float pre0 = gate * (t0s[k] + b0r[k]);
      float s0n  = fminf(fmaxf(0.5f * s0r[k] + 0.5f * pre0, 0.f), 1.f);
      float bn   = (s0n + p0r[k] >= 0.5f) ? 1.f : 0.f;   // uses OLD phi0
      p0r[k] = p0r[k] + s0n - bn;
      s0r[k] = s0n;
    }
    __syncthreads();   // protect red/t0s reuse next step
  }

#pragma unroll
  for (int i = 0; i < 8; ++i) {
    size_t g = (size_t)row * N1_ + tid + NTH2 * i;
    o_s1[g] = s1r[i]; o_phi1[g] = p1r[i];
  }
  if (tid < N0_) {
    o_s0[row * N0_ + tid]   = s0r[tid];
    o_phi0[row * N0_ + tid] = p0r[tid];
  }
}

// ---------------------------------------------------------------------------
extern "C" void kernel_launch(void* const* d_in, const int* in_sizes, int n_in,
                              void* d_out, int out_size, void* d_ws, size_t ws_size,
                              hipStream_t stream)
{
  (void)in_sizes; (void)n_in; (void)out_size; (void)d_ws; (void)ws_size;
  const float* s0   = (const float*)d_in[0];
  const float* s1   = (const float*)d_in[1];
  const float* x    = (const float*)d_in[2];
  const float* phi0 = (const float*)d_in[3];
  const float* phi1 = (const float*)d_in[4];
  const float* W0   = (const float*)d_in[5];
  const float* b0   = (const float*)d_in[6];
  const float* W1   = (const float*)d_in[7];
  const float* b1   = (const float*)d_in[8];

  float* out    = (float*)d_out;
  float* o_s0   = out;
  float* o_s1   = out + (size_t)B_ * N0_;
  float* o_phi0 = o_s1 + (size_t)B_ * N1_;
  float* o_phi1 = o_phi0 + (size_t)B_ * N0_;

  // Scratch staging inside d_out (safe: strictly ordered kernels on `stream`):
  //  - f16 padded copies of x and W1 live in the phi1 slot (67 MB; needs 13 MB)
  //    and are fully consumed by the GEMM before ep_relax writes phi1.
  //  - xW1 lives in the s1 slot and is consumed into registers by ep_relax
  //    before it writes the final s1 there.
  _Float16* xh  = (_Float16*)o_phi1;
  _Float16* wh  = xh + (size_t)B_ * KP_;
  float*    xW1 = o_s1;

  convert_f16<<<dim3(B_), 256, 0, stream>>>(x, W1, xh, wh);

  const int jobs = (B_ / 32) * (N1_ / 32);        // 16384 waves
  xw1_gemm_f16<<<dim3(jobs / 8), 256, 0, stream>>>(xh, wh, b1, xW1);

  const size_t smem = (size_t)(N0_ * N1_ + NW2 * N0_ + N0_) * sizeof(float);
  ep_relax<<<dim3(B_), NTH2, smem, stream>>>(s0, s1, phi0, phi1, W0, b0, xW1,
                                             o_s0, o_s1, o_phi0, o_phi1);
}
